// GroupQueryAttention_16776142258245
// MI455X (gfx1250) — compile-verified, matrix operand order checked by reading, not run on hardware
//
#include <hip/hip_runtime.h>
#include <math.h>

typedef __attribute__((ext_vector_type(16))) __bf16 v16bf;
typedef __attribute__((ext_vector_type(8)))  __bf16 v8bf;
typedef __attribute__((ext_vector_type(8)))  float  v8f;
typedef __attribute__((ext_vector_type(4)))  float  v4f;

#define D_MODEL  2048
#define KV_DIM   512
#define HEAD_DIM 128
#define N_GROUPS 4
#define HPG      4
#define BATCH    2
#define SEQ      2048
#define MTOT     (BATCH*SEQ)   // 4096 rows for all projections

// ======================= CDNA5 helpers =======================
// 16-bit A-fragment 16x32: lane<16 holds row (lane&15), K elems
// {kb..kb+7, kb+16..kb+23}, kb=(lane>>4)*8. B-fragment mirrors with
// col=lane&15. K-contiguous data => two 16B loads per fragment.
static __device__ __forceinline__ v16bf ld_frag16(const __bf16* p) {
  v8bf lo = *(const v8bf*)(p);
  v8bf hi = *(const v8bf*)(p + 16);
  v16bf r;
#pragma unroll
  for (int i = 0; i < 8; ++i) { r[i] = lo[i]; r[8 + i] = hi[i]; }
  return r;
}
static __device__ __forceinline__ v8f wmma_bf16(v16bf a, v16bf b, v8f c) {
  return __builtin_amdgcn_wmma_f32_16x16x32_bf16(false, a, false, b, (short)0, c,
                                                 false, false);
}
// Async global->LDS copy, 16B per lane. VDST = per-lane LDS offset
// (low 32 bits of the flat LDS address are the wave-relative offset).
static __device__ __forceinline__ void async_copy_b128(const __bf16* gsrc,
                                                       __bf16* ldst) {
  unsigned loff = (unsigned)(uintptr_t)ldst;
  unsigned long long gaddr = (unsigned long long)(uintptr_t)gsrc;
  asm volatile("global_load_async_to_lds_b128 %0, %1, off"
               :: "v"(loff), "v"(gaddr) : "memory");
}
static __device__ __forceinline__ void wait_async() {
  asm volatile("s_wait_asynccnt 0x0" ::: "memory");
}
// Wave-local LDS fence: DS ops are in-order per wave; this just stops the
// compiler reordering and drains DScnt for cross-lane RAW through LDS.
static __device__ __forceinline__ void wave_lds_fence() {
  asm volatile("s_wait_dscnt 0x0" ::: "memory");
}

// ================= weight pre-transpose: W[K,N] f32 -> Wt[N,K] bf16 ===========
__global__ __launch_bounds__(256) void transpose_w_kernel(
    const float* __restrict__ W, __bf16* __restrict__ Wt, int K, int N)
{
  __shared__ float t[32][33];
  const int tid = threadIdx.x;
  const int tx = tid & 31, ty = tid >> 5;        // 32 x 8
  const int k0 = blockIdx.x * 32, n0 = blockIdx.y * 32;
#pragma unroll
  for (int i = 0; i < 32; i += 8)
    t[ty + i][tx] = W[(size_t)(k0 + ty + i) * N + n0 + tx];
  __syncthreads();
#pragma unroll
  for (int i = 0; i < 32; i += 8)
    Wt[(size_t)(n0 + ty + i) * K + k0 + tx] = (__bf16)t[tx][ty + i];
}

// ======================= GEMM =======================
// C[M,N] = A[M,K] @ Wt[N,K]^T + bias.  Block tile 128x128, BK=32,
// 8 waves each computing 32x64 (8 WMMAs per step from 6 fragment loads).
// Double-buffered LDS; W tiles (and bf16 A tiles) staged with async-to-LDS,
// f32 A tiles staged via registers + packed cvt, committed after the WMMAs.
#define BM 128
#define BN 128
#define BK 32

struct ARegs { v4f f0, f1, f2, f3; };

static __device__ __forceinline__ void a_stage_issue(const float* g, __bf16* d,
                                                     ARegs& r) {
  (void)d;
  r.f0 = *(const v4f*)(g);
  r.f1 = *(const v4f*)(g + 4);
  r.f2 = *(const v4f*)(g + 8);
  r.f3 = *(const v4f*)(g + 12);
}
static __device__ __forceinline__ void a_stage_issue(const __bf16* g, __bf16* d,
                                                     ARegs& r) {
  (void)r;
  async_copy_b128(g, d);
  async_copy_b128(g + 8, d + 8);
}
// commit is a no-op for the bf16/async path
static __device__ __forceinline__ void a_commit_dispatch(__bf16* d, const ARegs& r,
                                                         const float*) {
#pragma unroll
  for (int i = 0; i < 4; ++i) {
    d[i]      = (__bf16)r.f0[i];
    d[4 + i]  = (__bf16)r.f1[i];
    d[8 + i]  = (__bf16)r.f2[i];
    d[12 + i] = (__bf16)r.f3[i];
  }
}
static __device__ __forceinline__ void a_commit_dispatch(__bf16*, const ARegs&,
                                                         const __bf16*) {}

static __device__ __forceinline__ void st_out(float* C, size_t i, float v)  { C[i] = v; }
static __device__ __forceinline__ void st_out(__bf16* C, size_t i, float v) { C[i] = (__bf16)v; }

template <typename AT, typename OT>
__global__ __launch_bounds__(256) void gemm_bias_kernel(
    const AT* __restrict__ A,       // [M,K] activations (f32 or bf16)
    const __bf16* __restrict__ Wt,  // [N,K] pre-transposed bf16 weights
    const float* __restrict__ bias, // [N]
    OT* __restrict__ C,             // [M,N]
    int N, int K)
{
  __shared__ __bf16 lA[2][BM * BK];   // [row][k]
  __shared__ __bf16 lB[2][BN * BK];   // [n][k]

  const int tid  = threadIdx.x;
  const int lane = tid & 31;
  const int w    = tid >> 5;
  const int m0   = blockIdx.x * BM;
  const int n0   = blockIdx.y * BN;

  const int mw = (w & 3) * 32;     // wave row band (32 rows)
  const int nw = (w >> 2) * 64;    // wave col band (64 cols)
  const int r   = lane & 15;
  const int hi2 = lane >> 4;
  const int kb  = hi2 * 8;

  // staging map: each thread owns 16 contiguous K-elements of one row
  const int srow = tid >> 1;            // 0..127
  const int sseg = (tid & 1) * 16;      // 0 or 16

  v8f acc[2][4];
#pragma unroll
  for (int ms = 0; ms < 2; ++ms)
#pragma unroll
    for (int ns = 0; ns < 4; ++ns) acc[ms][ns] = (v8f){0.f,0.f,0.f,0.f,0.f,0.f,0.f,0.f};

  const int nsteps = K / BK;

  // prologue: stage step 0 into buffer 0
  {
    ARegs ar;
    a_stage_issue(A + (size_t)(m0 + srow) * K + sseg, &lA[0][srow * BK + sseg], ar);
    const __bf16* wg = Wt + (size_t)(n0 + srow) * K + sseg;
    __bf16* wd = &lB[0][srow * BK + sseg];
    async_copy_b128(wg, wd);
    async_copy_b128(wg + 8, wd + 8);
    a_commit_dispatch(&lA[0][srow * BK + sseg], ar, A);
  }
  wait_async();
  __syncthreads();

  for (int s = 0; s < nsteps; ++s) {
    const int cur = s & 1;
    const int nxt = cur ^ 1;
    const int k1  = (s + 1) * BK;

    ARegs ar;
    __bf16* ad = &lA[nxt][srow * BK + sseg];
    if (s + 1 < nsteps) {
      // issue next tile: W via async-to-LDS, A via regs (f32) or async (bf16)
      const __bf16* wg = Wt + (size_t)(n0 + srow) * K + k1 + sseg;
      __bf16* wd = &lB[nxt][srow * BK + sseg];
      async_copy_b128(wg, wd);
      async_copy_b128(wg + 8, wd + 8);
      a_stage_issue(A + (size_t)(m0 + srow) * K + k1 + sseg, ad, ar);
    }

    // compute current tile: 2 A-frags, 4 B-frags, 8 WMMAs
    v16bf af0 = ld_frag16(&lA[cur][(mw + r) * BK + kb]);
    v16bf af1 = ld_frag16(&lA[cur][(mw + 16 + r) * BK + kb]);
#pragma unroll
    for (int ns = 0; ns < 4; ++ns) {
      v16bf bf = ld_frag16(&lB[cur][(nw + ns * 16 + r) * BK + kb]);
      acc[0][ns] = wmma_bf16(af0, bf, acc[0][ns]);
      acc[1][ns] = wmma_bf16(af1, bf, acc[1][ns]);
    }

    if (s + 1 < nsteps) a_commit_dispatch(ad, ar, A);   // f32 path: cvt+store now
    wait_async();
    __syncthreads();
  }

  // epilogue: bias + store
#pragma unroll
  for (int ms = 0; ms < 2; ++ms) {
    const int rbase = m0 + mw + ms * 16 + 8 * hi2;
#pragma unroll
    for (int ns = 0; ns < 4; ++ns) {
      const int col = n0 + nw + ns * 16 + r;
      const float bb = bias[col];
#pragma unroll
      for (int i = 0; i < 8; ++i)
        st_out(C, (size_t)(rbase + i) * N + col, acc[ms][ns][i] + bb);
    }
  }
}

// ======================= GQA flash attention (all bf16 I/O) =======================
// Block = 8 waves = 4 heads x 2 q-tiles sharing one KV group's tiles.
#define KVT 32

__global__ __launch_bounds__(256) void gqa_attn_kernel(
    const __bf16* __restrict__ QP,   // [B*S, 2048]
    const __bf16* __restrict__ KP,   // [B*S, 512]
    const __bf16* __restrict__ VP,   // [B*S, 512]
    __bf16* __restrict__ OUT)        // [B*S, 2048]
{
  __shared__ __bf16 lK[KVT * HEAD_DIM];    // [j][d] row-major
  __shared__ __bf16 lVt[HEAD_DIM * KVT];   // [d][j] transposed
  __shared__ float  lS[8][16 * KVT];       // per-wave score scratch
  __shared__ __bf16 lP[8][16 * KVT];       // per-wave probabilities
  __shared__ float  lRow[8][16];           // per-wave alpha / 1/l

  const int tid = threadIdx.x, lane = tid & 31, w = tid >> 5;
  const int bid   = blockIdx.x;
  const int qpair = bid & 63;              // S/16/2
  const int g     = (bid >> 6) & 3;
  const int b     = bid >> 8;
  const int h     = w & 3;
  const int qt    = qpair * 2 + (w >> 2);

  const int r   = lane & 15;
  const int hi2 = lane >> 4;
  const int kb  = hi2 * 8;
  const int headcol = (g * HPG + h) * HEAD_DIM;

  // Q tile (16 x 128) preloaded as 4 A-fragments straight from bf16 global.
  v16bf qf[4];
  {
    const __bf16* qbase = QP + (size_t)(b * SEQ + qt * 16 + r) * D_MODEL + headcol + kb;
#pragma unroll
    for (int c = 0; c < 4; ++c) qf[c] = ld_frag16(qbase + c * 32);
  }

  v8f zero8 = {0.f,0.f,0.f,0.f,0.f,0.f,0.f,0.f};
  v8f o[8];
#pragma unroll
  for (int t = 0; t < 8; ++t) o[t] = zero8;
  float m = -3.0e38f, l = 0.f;               // lanes 0..15 only
  const float scale = 0.08838834764831845f;  // 1/sqrt(128)

  const int sj   = tid >> 3;                 // staged kv row 0..31
  const int dseg = (tid & 7) * 16;           // 0..112

  for (int kv0 = 0; kv0 < SEQ; kv0 += KVT) {
    __syncthreads();                         // prior tile fully consumed
    {
      const size_t kvrow = (size_t)(b * SEQ + kv0 + sj) * KV_DIM + g * HEAD_DIM + dseg;
      // K tile: pure async copy (bf16, K-contiguous rows)
      const __bf16* ksrc = KP + kvrow;
      __bf16* kdst = &lK[sj * HEAD_DIM + dseg];
      async_copy_b128(ksrc, kdst);
      async_copy_b128(ksrc + 8, kdst + 8);
      // V tile: transpose while staging
      const __bf16* vsrc = VP + kvrow;
      v8bf v0 = *(const v8bf*)(vsrc);
      v8bf v1 = *(const v8bf*)(vsrc + 8);
#pragma unroll
      for (int e = 0; e < 8; ++e) {
        lVt[(dseg + e) * KVT + sj]     = v0[e];
        lVt[(dseg + 8 + e) * KVT + sj] = v1[e];
      }
    }
    wait_async();
    __syncthreads();

    // S = Q K^T : 16x32 tile, reduce d in 4 chunks of 32 -> 8 WMMAs
    v8f s0 = zero8, s1 = zero8;
#pragma unroll
    for (int c = 0; c < 4; ++c) {
      v16bf k0 = ld_frag16(&lK[(r)      * HEAD_DIM + c * 32 + kb]);
      v16bf k1 = ld_frag16(&lK[(16 + r) * HEAD_DIM + c * 32 + kb]);
      s0 = wmma_bf16(qf[c], k0, s0);
      s1 = wmma_bf16(qf[c], k1, s1);
    }
    {
      float* Sw = &lS[w][0];
#pragma unroll
      for (int i = 0; i < 8; ++i) {
        int row = i + 8 * hi2;
        Sw[row * KVT + r]      = s0[i] * scale;
        Sw[row * KVT + 16 + r] = s1[i] * scale;
      }
    }
    wave_lds_fence();   // wave-private scratch: no block barrier needed

    // online softmax, lanes 0..15 own one query row each
    if (lane < 16) {
      const float* Sw = &lS[w][lane * KVT];
      float cmax = Sw[0];
#pragma unroll
      for (int j = 1; j < KVT; ++j) cmax = fmaxf(cmax, Sw[j]);
      float mnew  = fmaxf(m, cmax);
      float alpha = __expf(m - mnew);
      float sum = 0.f;
      __bf16* Pw = &lP[w][lane * KVT];
#pragma unroll
      for (int j = 0; j < KVT; ++j) {
        float p = __expf(Sw[j] - mnew);
        sum += p;
        Pw[j] = (__bf16)p;
      }
      l = l * alpha + sum;
      m = mnew;
      lRow[w][lane] = alpha;
    }
    wave_lds_fence();

    // rescale O, then O += P V (8 d-tiles) -> 8 WMMAs
    {
      v4f a0 = *(const v4f*)&lRow[w][8 * hi2];
      v4f a1 = *(const v4f*)&lRow[w][8 * hi2 + 4];
#pragma unroll
      for (int t = 0; t < 8; ++t) {
#pragma unroll
        for (int i = 0; i < 4; ++i) { o[t][i] *= a0[i]; o[t][4 + i] *= a1[i]; }
      }
      v16bf pf = ld_frag16(&lP[w][r * KVT + kb]);
#pragma unroll
      for (int t = 0; t < 8; ++t) {
        v16bf vf = ld_frag16(&lVt[(t * 16 + r) * KVT + kb]);
        o[t] = wmma_bf16(pf, vf, o[t]);
      }
    }
  }

  if (lane < 16) lRow[w][lane] = 1.0f / l;
  wave_lds_fence();
  {
    v4f i0 = *(const v4f*)&lRow[w][8 * hi2];
    v4f i1 = *(const v4f*)&lRow[w][8 * hi2 + 4];
    const int rbase = b * SEQ + qt * 16 + 8 * hi2;
#pragma unroll
    for (int t = 0; t < 8; ++t) {
      const int col = headcol + t * 16 + r;
#pragma unroll
      for (int i = 0; i < 8; ++i) {
        float inv = (i < 4) ? i0[i] : i1[i - 4];
        OUT[(size_t)(rbase + i) * D_MODEL + col] = (__bf16)(o[t][i] * inv);
      }
    }
  }
}

// ======================= launcher =======================
extern "C" void kernel_launch(void* const* d_in, const int* in_sizes, int n_in,
                              void* d_out, int out_size, void* d_ws, size_t ws_size,
                              hipStream_t stream) {
  (void)in_sizes; (void)n_in; (void)out_size; (void)ws_size;
  const float* q  = (const float*)d_in[0];
  const float* k  = (const float*)d_in[1];
  const float* v  = (const float*)d_in[2];
  const float* Wq = (const float*)d_in[3];
  const float* bq = (const float*)d_in[4];
  const float* Wk = (const float*)d_in[5];
  const float* bk = (const float*)d_in[6];
  const float* Wv = (const float*)d_in[7];
  const float* bv = (const float*)d_in[8];
  const float* Wc = (const float*)d_in[9];
  const float* bc = (const float*)d_in[10];
  float* out = (float*)d_out;

  // workspace (all bf16): qp kp vp attn | Wtq Wtk Wtv Wtc  (~60 MB)
  __bf16* qp   = (__bf16*)d_ws;
  __bf16* kp   = qp   + (size_t)MTOT * D_MODEL;
  __bf16* vp   = kp   + (size_t)MTOT * KV_DIM;
  __bf16* attn = vp   + (size_t)MTOT * KV_DIM;
  __bf16* Wtq  = attn + (size_t)MTOT * D_MODEL;
  __bf16* Wtk  = Wtq  + (size_t)D_MODEL * D_MODEL;
  __bf16* Wtv  = Wtk  + (size_t)D_MODEL * KV_DIM;
  __bf16* Wtc  = Wtv  + (size_t)D_MODEL * KV_DIM;

  dim3 blk(256);
  // weight pre-transpose/convert (f32 [K,N] -> bf16 [N,K])
  transpose_w_kernel<<<dim3(D_MODEL/32, D_MODEL/32), blk, 0, stream>>>(Wq, Wtq, D_MODEL, D_MODEL);
  transpose_w_kernel<<<dim3(D_MODEL/32, KV_DIM /32), blk, 0, stream>>>(Wk, Wtk, D_MODEL, KV_DIM);
  transpose_w_kernel<<<dim3(D_MODEL/32, KV_DIM /32), blk, 0, stream>>>(Wv, Wtv, D_MODEL, KV_DIM);
  transpose_w_kernel<<<dim3(D_MODEL/32, D_MODEL/32), blk, 0, stream>>>(Wc, Wtc, D_MODEL, D_MODEL);

  // projections (f32 activations -> bf16 outputs)
  gemm_bias_kernel<float, __bf16><<<dim3(MTOT/BM, D_MODEL/BN), blk, 0, stream>>>(q, Wtq, bq, qp, D_MODEL, D_MODEL);
  gemm_bias_kernel<float, __bf16><<<dim3(MTOT/BM, KV_DIM /BN), blk, 0, stream>>>(k, Wtk, bk, kp, KV_DIM,  D_MODEL);
  gemm_bias_kernel<float, __bf16><<<dim3(MTOT/BM, KV_DIM /BN), blk, 0, stream>>>(v, Wtv, bv, vp, KV_DIM,  D_MODEL);

  // attention (bf16 in/out, f32 softmax + accumulation)
  gqa_attn_kernel<<<dim3(BATCH * N_GROUPS * (SEQ/16/2)), blk, 0, stream>>>(qp, kp, vp, attn);

  // output projection (bf16 activations -> f32 result)
  gemm_bias_kernel<__bf16, float><<<dim3(MTOT/BM, D_MODEL/BN), blk, 0, stream>>>(attn, Wtc, bc, out, D_MODEL, D_MODEL);
}
